// GRCN_75574244540609
// MI455X (gfx1250) — matrix operand order repaired and checked by reading, not compiled, not run on hardware
//
#include <hip/hip_runtime.h>

// ---------------------------------------------------------------------------
// Types for CDNA5 WMMA (wave32): v_wmma_f32_16x16x32_bf16
// ---------------------------------------------------------------------------
typedef __attribute__((ext_vector_type(16))) __bf16 v16bf;
typedef __attribute__((ext_vector_type(8)))  float  v8f;

union BF16x16 { unsigned int d[8]; v16bf v; };
union F32x8   { float f[8];        v8f   v; };

__device__ __forceinline__ unsigned short f2bf(float x) {
    // round-to-nearest-even float -> bf16
    unsigned int u = __float_as_uint(x);
    u += 0x7FFFu + ((u >> 16) & 1u);
    return (unsigned short)(u >> 16);
}
__device__ __forceinline__ unsigned int pack2(float a, float b) {
    return (unsigned int)f2bf(a) | ((unsigned int)f2bf(b) << 16);
}

// ---------------------------------------------------------------------------
// bf16-input / f32-accumulate WMMA GEMM: C[MxN] = A[MxK] * B
//   BT==0 : B is row-major KxN          BT==1 : B is row-major NxK (A * B^T)
//   GUARD : 1 only when Ncol may be < 128 (the C=16 GEMMs)
// Assumptions (true for every call site): M % 128 == 0, K % 64 == 0,
// GUARD==0 implies Ncol % 128 == 0. Block tile 128x128, BK=64,
// 256 threads = 8 waves, each wave owns a 32x64 sub-tile (8 accumulators).
// ---------------------------------------------------------------------------
template <int BT, int GUARD>
__global__ __launch_bounds__(256)
void wmma_gemm(const float* __restrict__ A, const float* __restrict__ B,
               float* __restrict__ C, int M, int Ncol, int K)
{
    __shared__ unsigned int sA[128][36];  // 128 x 64 bf16 (32 dw) + 4 dw pad
    __shared__ unsigned int sB[64][70];   // 64 x 128 bf16 (64 dw) + 6 dw pad

    const int tid  = threadIdx.x;
    const int lane = tid & 31;
    const int wave = tid >> 5;
    const int wm   = wave >> 1;     // 0..3 -> row offset wm*32
    const int wn   = wave & 1;      // 0..1 -> col offset wn*64
    const int bm   = blockIdx.y * 128;
    const int bn   = blockIdx.x * 128;

    const int lda = K;
    const int ldb = BT ? K : Ncol;

    v8f acc[2][4] = {};

    const int arow0 = wm * 32 + (lane & 15);
    const int kb4   = (lane >> 4) * 4;                 // A: dword offset of K-half
    const int brow0 = (lane & 15) + ((lane >> 4) * 16);// B: K row within 32-step

    for (int kk = 0; kk < K; kk += 64) {
        // ---- prefetch next A K-tile (global_prefetch_b8) ----
        if (kk + 64 < K)
            __builtin_prefetch(A + (size_t)(bm + (tid >> 1)) * lda + kk + 64 +
                               (tid & 1) * 32, 0, 3);

        // ---- stage A tile 128x64: float4 loads, dword-packed bf16 stores ----
        #pragma unroll
        for (int it = 0; it < 8; it++) {
            int i  = tid + it * 256;
            int m  = i >> 4;              // 16 float4 per 64-wide row
            int c4 = (i & 15) * 4;
            const float4 f = *(const float4*)(A + (size_t)(bm + m) * lda + kk + c4);
            sA[m][(c4 >> 1) + 0] = pack2(f.x, f.y);
            sA[m][(c4 >> 1) + 1] = pack2(f.z, f.w);
        }
        // ---- stage B tile 64x128 ----
        if (BT) {            // B row-major NxK: contiguous along K, scatter into sB[k][n]
            unsigned short* p = (unsigned short*)sB;   // row stride = 140 ushorts
            #pragma unroll
            for (int it = 0; it < 8; it++) {
                int i  = tid + it * 256;
                int n  = i >> 4;
                int c4 = (i & 15) * 4;
                const float4 f = *(const float4*)(B + (size_t)(bn + n) * ldb + kk + c4);
                p[(c4 + 0) * 140 + n] = f2bf(f.x);
                p[(c4 + 1) * 140 + n] = f2bf(f.y);
                p[(c4 + 2) * 140 + n] = f2bf(f.z);
                p[(c4 + 3) * 140 + n] = f2bf(f.w);
            }
        } else if (!GUARD) { // B row-major KxN, full 128-wide tile
            #pragma unroll
            for (int it = 0; it < 8; it++) {
                int i  = tid + it * 256;
                int k  = i >> 5;              // 32 float4 per 128-wide row
                int c4 = (i & 31) * 4;
                const float4 f = *(const float4*)(B + (size_t)(kk + k) * ldb + bn + c4);
                sB[k][(c4 >> 1) + 0] = pack2(f.x, f.y);
                sB[k][(c4 >> 1) + 1] = pack2(f.z, f.w);
            }
        } else {             // narrow-N path (Ncol may be 16): scalar guarded
            unsigned short* p = (unsigned short*)sB;
            for (int i = tid; i < 64 * 128; i += 256) {
                int k = i >> 7, n = i & 127;
                float v = (bn + n < Ncol) ? B[(size_t)(kk + k) * ldb + bn + n] : 0.f;
                p[k * 140 + n] = f2bf(v);
            }
        }
        __syncthreads();

        // ---- two WMMA K-steps of 32 ----
        #pragma unroll
        for (int s = 0; s < 2; s++) {
            v16bf afrag[2];
            #pragma unroll
            for (int mt = 0; mt < 2; mt++) {
                BF16x16 t;
                int m = arow0 + mt * 16;
                #pragma unroll
                for (int v = 0; v < 8; v++) {
                    int kd = ((v < 4) ? v : (v + 4)) + kb4;   // ISA A-frag K order
                    t.d[v] = sA[m][s * 16 + kd];
                }
                afrag[mt] = t.v;
            }
            v16bf bfrag[4];
            #pragma unroll
            for (int nt = 0; nt < 4; nt++) {
                BF16x16 t;
                const unsigned short* pb = (const unsigned short*)sB;
                int n   = wn * 64 + nt * 16 + (lane & 15);
                #pragma unroll
                for (int v = 0; v < 8; v++) {
                    int k0 = s * 32 + ((v < 4) ? 2 * v : 2 * v + 8) + ((lane >> 4) * 8);
                    t.d[v] = (unsigned int)pb[k0 * 140 + n] | ((unsigned int)pb[(k0 + 1) * 140 + n] << 16);
                }
                bfrag[nt] = t.v;
            }
            #pragma unroll
            for (int mt = 0; mt < 2; mt++)
                #pragma unroll
                for (int nt = 0; nt < 4; nt++)
                    acc[mt][nt] = __builtin_amdgcn_wmma_f32_16x16x32_bf16(
                        false, afrag[mt], false, bfrag[nt],
                        (short)0, acc[mt][nt], false, false);
        }
        __syncthreads();
    }

    // ---- store C (C/D layout: lane = N, VGPR v = M row, +8 for upper lanes) ----
    #pragma unroll
    for (int mt = 0; mt < 2; mt++) {
        #pragma unroll
        for (int nt = 0; nt < 4; nt++) {
            F32x8 t; t.v = acc[mt][nt];
            int gn = bn + wn * 64 + nt * 16 + (lane & 15);
            if (GUARD && gn >= Ncol) continue;
            #pragma unroll
            for (int v = 0; v < 8; v++) {
                int gm = bm + wm * 32 + mt * 16 + v + ((lane >> 4) * 8);
                C[(size_t)gm * Ncol + gn] = t.f[v];
            }
        }
    }
}

// ---------------------------------------------------------------------------
// Sparse graph helpers
// ---------------------------------------------------------------------------
__global__ void deg_k(float* __restrict__ deg, const int* __restrict__ dst,
                      const float* __restrict__ w, int E) {
    int e = blockIdx.x * blockDim.x + threadIdx.x;
    if (e < E) atomicAdd(&deg[dst[e]], w[e]);
}

__global__ void dinv_k(float* __restrict__ dinv, const float* __restrict__ deg, int N) {
    int i = blockIdx.x * blockDim.x + threadIdx.x;
    if (i < N) { float d = deg[i]; dinv[i] = (d > 0.f) ? rsqrtf(d) : 0.f; }
}

__global__ void wn_k(float* __restrict__ wn, const float* __restrict__ w,
                     const float* __restrict__ dinv, const int* __restrict__ src,
                     const int* __restrict__ dst, int E) {
    int e = blockIdx.x * blockDim.x + threadIdx.x;
    if (e < E) wn[e] = w[e] * dinv[src[e]] * dinv[dst[e]];
}

// Y[dst] += w * X[src]   (one block per edge)
__global__ void spmm_k(const int* __restrict__ src, const int* __restrict__ dst,
                       const float* __restrict__ w, const float* __restrict__ X,
                       float* __restrict__ Y, int E, int Fd) {
    int e = blockIdx.x;
    int s = src[e], d = dst[e];
    float ww = w[e];
    const float* xs = X + (size_t)s * Fd;
    float*       yd = Y + (size_t)d * Fd;
    for (int f = threadIdx.x; f < Fd; f += blockDim.x)
        atomicAdd(&yd[f], ww * xs[f]);
}

__global__ void add_edges_k(float* __restrict__ A, const int* __restrict__ src,
                            const int* __restrict__ dst, const float* __restrict__ w,
                            int E, int N) {
    int e = blockIdx.x * blockDim.x + threadIdx.x;
    if (e < E) atomicAdd(&A[(size_t)dst[e] * N + src[e]], w[e]);
}

// ---------------------------------------------------------------------------
// Elementwise epilogues
// ---------------------------------------------------------------------------
__global__ void relu_bias_k(float* __restrict__ X, const float* __restrict__ b,
                            int total, int Fd) {
    int idx = blockIdx.x * blockDim.x + threadIdx.x;
    if (idx < total) X[idx] = fmaxf(X[idx] + b[idx % Fd], 0.f);
}

__global__ void scale_rows_k(float* __restrict__ X, const float* __restrict__ di,
                             int total, int Fd) {
    int idx = blockIdx.x * blockDim.x + threadIdx.x;
    if (idx < total) X[idx] *= di[idx / Fd];
}

__global__ void scale_bias_relu_k(float* __restrict__ X, const float* __restrict__ di,
                                  const float* __restrict__ b, int total, int Fd) {
    int idx = blockIdx.x * blockDim.x + threadIdx.x;
    if (idx < total) X[idx] = fmaxf(di[idx / Fd] * X[idx] + b[idx % Fd], 0.f);
}

__global__ void final_out_k(float* __restrict__ out, const float* __restrict__ t,
                            const float* __restrict__ di, const float* __restrict__ b,
                            int total, int Cd) {
    int idx = blockIdx.x * blockDim.x + threadIdx.x;
    if (idx < total) out[idx] = di[idx / Cd] * t[idx] + b[idx % Cd];
}

// emb[i,:] = (emb[i,:] + b) / clip(||emb[i,:]+b||, 1e-12)   (Fd <= 512)
__global__ __launch_bounds__(256)
void bias_l2norm_k(float* __restrict__ X, const float* __restrict__ b, int Fd) {
    __shared__ float buf[512];
    __shared__ float red[256];
    int i = blockIdx.x, tid = threadIdx.x;
    float s = 0.f;
    for (int f = tid; f < Fd; f += 256) {
        float v = X[(size_t)i * Fd + f] + b[f];
        buf[f] = v; s += v * v;
    }
    red[tid] = s; __syncthreads();
    for (int st = 128; st > 0; st >>= 1) {
        if (tid < st) red[tid] += red[tid + st];
        __syncthreads();
    }
    float sc = 1.0f / fmaxf(sqrtf(red[0]), 1e-12f);
    for (int f = tid; f < Fd; f += 256) X[(size_t)i * Fd + f] = buf[f] * sc;
}

__global__ __launch_bounds__(256)
void rowsum_di_k(const float* __restrict__ A, float* __restrict__ di, int N) {
    __shared__ float red[256];
    int i = blockIdx.x, tid = threadIdx.x;
    float s = 0.f;
    for (int j = tid; j < N; j += 256) s += A[(size_t)i * N + j];
    red[tid] = s; __syncthreads();
    for (int st = 128; st > 0; st >>= 1) {
        if (tid < st) red[tid] += red[tid + st];
        __syncthreads();
    }
    if (tid == 0) { float d = red[0]; di[i] = (d > 0.f) ? rsqrtf(d) : 0.f; }
}

// ---------------------------------------------------------------------------
// Top-64 per row, in place: keep top-K values at their columns, zero the rest.
// ---------------------------------------------------------------------------
__global__ __launch_bounds__(256)
void topk_k(float* __restrict__ A, int N) {
    __shared__ float row[8192];
    __shared__ float rv[256];
    __shared__ int   ri[256];
    __shared__ float tv[64];
    __shared__ int   ti[64];
    const int tid = threadIdx.x;
    float* g = A + (size_t)blockIdx.x * N;
    for (int j = tid; j < N; j += 256) row[j] = g[j];
    __syncthreads();
    for (int t = 0; t < 64; t++) {
        float bv = -3.4e38f; int bi = 0x7fffffff;
        for (int j = tid; j < N; j += 256) {
            float v = row[j];
            if (v > bv) { bv = v; bi = j; }
        }
        rv[tid] = bv; ri[tid] = bi;
        __syncthreads();
        for (int st = 128; st > 0; st >>= 1) {
            if (tid < st) {
                if (rv[tid + st] > rv[tid] ||
                    (rv[tid + st] == rv[tid] && ri[tid + st] < ri[tid])) {
                    rv[tid] = rv[tid + st]; ri[tid] = ri[tid + st];
                }
            }
            __syncthreads();
        }
        if (tid == 0) { tv[t] = rv[0]; ti[t] = ri[0]; row[ri[0]] = -3.4e38f; }
        __syncthreads();
    }
    for (int j = tid; j < N; j += 256) g[j] = 0.f;
    __syncthreads();
    if (tid < 64) g[ti[tid]] = tv[tid];
}

// in-place A = 0.5*(A + A^T)
__global__ void symmetrize_k(float* __restrict__ A, int N) {
    long long idx = (long long)blockIdx.x * blockDim.x + threadIdx.x;
    if (idx >= (long long)N * N) return;
    int i = (int)(idx / N), j = (int)(idx % N);
    if (i < j) {
        float a = A[(size_t)i * N + j], b = A[(size_t)j * N + i];
        float m = 0.5f * (a + b);
        A[(size_t)i * N + j] = m;
        A[(size_t)j * N + i] = m;
    }
}

// ---------------------------------------------------------------------------
// Host orchestration
// ---------------------------------------------------------------------------
extern "C" void kernel_launch(void* const* d_in, const int* in_sizes, int n_in,
                              void* d_out, int out_size, void* d_ws, size_t ws_size,
                              hipStream_t stream) {
    const int N = 8192, F = 512, HG = 512, DE = 512, HT = 256, Cc = 16, E = 131072;

    const float* x   = (const float*)d_in[0];
    const int*   ei  = (const int*)  d_in[1];
    const float* ew  = (const float*)d_in[2];
    const float* Wg1 = (const float*)d_in[3];
    const float* bg1 = (const float*)d_in[4];
    const float* Wg2 = (const float*)d_in[5];
    const float* bg2 = (const float*)d_in[6];
    const float* Wt1 = (const float*)d_in[7];
    const float* bt1 = (const float*)d_in[8];
    const float* Wt2 = (const float*)d_in[9];
    const float* bt2 = (const float*)d_in[10];
    const int* src = ei;        // edge_index[0]
    const int* dst = ei + E;    // edge_index[1]

    float* out  = (float*)d_out;                 // [N, C]
    float* Anew = out + (size_t)N * Cc;          // [N, N] (also the sim buffer)
    float* Afin = Anew + (size_t)N * N;          // [N, N]

    float* ws = (float*)d_ws;
    size_t off = 0;
    float* deg  = ws + off; off += N;
    float* dinv = ws + off; off += N;
    float* di   = ws + off; off += N;
    float* wn   = ws + off; off += E;
    float* bufA = ws + off; off += (size_t)N * 512;
    float* bufB = ws + off; off += (size_t)N * 512;
    float* bufC = ws + off; off += (size_t)N * 512;

    dim3 blk(256);

    // --- normalize_a (sparse): deg, dinv, w_n ---
    hipMemsetAsync(deg, 0, N * sizeof(float), stream);
    deg_k <<<(E + 255) / 256, blk, 0, stream>>>(deg, dst, ew, E);
    dinv_k<<<(N + 255) / 256, blk, 0, stream>>>(dinv, deg, N);
    wn_k  <<<(E + 255) / 256, blk, 0, stream>>>(wn, ew, dinv, src, dst, E);

    // --- conv_graph layer 1: h = relu(spmm(x @ Wg1) + bg1) ---
    wmma_gemm<0,0><<<dim3(HG / 128, N / 128), blk, 0, stream>>>(x, Wg1, bufA, N, HG, F);
    hipMemsetAsync(bufB, 0, (size_t)N * HG * sizeof(float), stream);
    spmm_k<<<E, 128, 0, stream>>>(src, dst, wn, bufA, bufB, E, HG);
    relu_bias_k<<<(N * HG + 255) / 256, blk, 0, stream>>>(bufB, bg1, N * HG, HG);

    // --- conv_graph layer 2: emb = spmm(h @ Wg2) + bg2, then row-L2 normalize ---
    wmma_gemm<0,0><<<dim3(DE / 128, N / 128), blk, 0, stream>>>(bufB, Wg2, bufA, N, DE, HG);
    hipMemsetAsync(bufC, 0, (size_t)N * DE * sizeof(float), stream);
    spmm_k<<<E, 128, 0, stream>>>(src, dst, wn, bufA, bufC, E, DE);
    bias_l2norm_k<<<N, blk, 0, stream>>>(bufC, bg2, DE);

    // --- sim = emb @ emb^T (half-split sum == full inner product) ---
    wmma_gemm<1,0><<<dim3(N / 128, N / 128), blk, 0, stream>>>(bufC, bufC, Anew, N, N, DE);

    // --- KNN(64) sparsify + symmetrize ---
    topk_k<<<N, blk, 0, stream>>>(Anew, N);
    symmetrize_k<<<(int)(((long long)N * N + 255) / 256), blk, 0, stream>>>(Anew, N);

    // --- fuse: A_final = A_new + scatter(edge_weight at (dst, src)) ---
    hipMemcpyAsync(Afin, Anew, (size_t)N * N * sizeof(float),
                   hipMemcpyDeviceToDevice, stream);
    add_edges_k<<<(E + 255) / 256, blk, 0, stream>>>(Afin, src, dst, ew, E, N);

    // --- normalize_a (dense): di = rowsum^-1/2; A_norm folded into GEMM operands ---
    rowsum_di_k<<<N, blk, 0, stream>>>(Afin, di, N);

    // --- conv_task layer 1: h2 = relu(di_i * (Afin @ (di_j * (x@Wt1))) + bt1) ---
    wmma_gemm<0,0><<<dim3(HT / 128, N / 128), blk, 0, stream>>>(x, Wt1, bufA, N, HT, F);
    scale_rows_k<<<(N * HT + 255) / 256, blk, 0, stream>>>(bufA, di, N * HT, HT);
    wmma_gemm<0,0><<<dim3(HT / 128, N / 128), blk, 0, stream>>>(Afin, bufA, bufB, N, HT, N);
    scale_bias_relu_k<<<(N * HT + 255) / 256, blk, 0, stream>>>(bufB, di, bt1, N * HT, HT);

    // --- conv_task layer 2: out = di_i * (Afin @ (di_j * (h2@Wt2))) + bt2 ---
    wmma_gemm<0,1><<<dim3((Cc + 127) / 128, N / 128), blk, 0, stream>>>(bufB, Wt2, bufA, N, Cc, HT);
    scale_rows_k<<<(N * Cc + 255) / 256, blk, 0, stream>>>(bufA, di, N * Cc, Cc);
    wmma_gemm<0,1><<<dim3((Cc + 127) / 128, N / 128), blk, 0, stream>>>(Afin, bufA, bufC, N, Cc, N);
    final_out_k<<<(N * Cc + 255) / 256, blk, 0, stream>>>(out, bufC, di, bt2, N * Cc, Cc);
}
